// GCNEncoder2_82781199663551
// MI455X (gfx1250) — matrix operand order repaired and checked by reading, not compiled, not run on hardware
//
#include <hip/hip_runtime.h>

typedef __attribute__((ext_vector_type(16))) __bf16          v16bf;
typedef __attribute__((ext_vector_type(16))) unsigned short  v16us;
typedef __attribute__((ext_vector_type(8)))  float           v8f;

#define LEAKY_SLOPE 0.01f
#define BN_EPS      1e-5f

__device__ __forceinline__ void bf16_split(float f, unsigned short &hi, unsigned short &lo) {
  unsigned u  = __float_as_uint(f);
  unsigned rh = (u + 0x7FFFu + ((u >> 16) & 1u)) & 0xFFFF0000u;   // RNE to bf16
  hi = (unsigned short)(rh >> 16);
  float    fl = f - __uint_as_float(rh);                          // exact residual
  unsigned ul = __float_as_uint(fl);
  lo = (unsigned short)((ul + 0x7FFFu + ((ul >> 16) & 1u)) >> 16);
}

// ---------------- utility ----------------
__global__ void k_zero(float *p, int n) {
  int i = blockIdx.x * blockDim.x + threadIdx.x;
  if (i < n) p[i] = 0.0f;
}

__global__ void k_degree(const int *__restrict__ dst, float *__restrict__ deg, int E) {
  int i = blockIdx.x * blockDim.x + threadIdx.x;
  if (i < E) atomicAdd(&deg[dst[i]], 1.0f);
}

__global__ void k_dinv(float *__restrict__ deg, int N) {
  int i = blockIdx.x * blockDim.x + threadIdx.x;
  if (i < N) deg[i] = rsqrtf(deg[i] + 1.0f);   // +1 = self loop
}

// W[k][n] (128x128 row-major) -> transposed bf16 hi/lo: T*[n*128 + k]
__global__ void k_prep_w(const float *__restrict__ W, unsigned short *__restrict__ Th,
                         unsigned short *__restrict__ Tl) {
  int idx = blockIdx.x * blockDim.x + threadIdx.x;
  if (idx >= 128 * 128) return;
  int k = idx >> 7, n = idx & 127;
  unsigned short h, l;
  bf16_split(W[idx], h, l);
  Th[n * 128 + k] = h;
  Tl[n * 128 + k] = l;
}

// ---------------- WMMA GEMM: Out[N x 128] = A[N x 128] * B(128x128) ----------------
// bf16x2 error-compensated: acc += Ah*Bh + Ah*Bl + Al*Bh
__global__ __launch_bounds__(256) void k_gemm128(const float *__restrict__ A,
                                                 const unsigned short *__restrict__ BTh,
                                                 const unsigned short *__restrict__ BTl,
                                                 float *__restrict__ Out, int nrows) {
  const int tid  = threadIdx.x;
  const int wave = tid >> 5;
  const int lane = tid & 31;
  const int half = lane >> 4;
  const int l15  = lane & 15;
  const int row0 = blockIdx.x * 128 + wave * 16;

  v8f acc[8] = {};

  int arow = row0 + l15;
  if (arow >= nrows) arow = nrows - 1;          // clamp: keep EXEC all-ones for WMMA
  const float *arp = A + (long)arow * 128;

  #pragma unroll
  for (int kc = 0; kc < 4; ++kc) {
    // A 16x32 layout (contiguous, matches B): lanes 0-15 elems0-15 = K 0..15;
    // lanes 16-31: K 16..31  (relative to kc*32)
    const int kb0 = kc * 32 + half * 16;
    const int kb1 = kb0 + 8;
    float4 p0 = *(const float4 *)(arp + kb0);
    float4 p1 = *(const float4 *)(arp + kb0 + 4);
    float4 p2 = *(const float4 *)(arp + kb1);
    float4 p3 = *(const float4 *)(arp + kb1 + 4);
    float va[16] = {p0.x, p0.y, p0.z, p0.w, p1.x, p1.y, p1.z, p1.w,
                    p2.x, p2.y, p2.z, p2.w, p3.x, p3.y, p3.z, p3.w};
    v16us ahu, alu;
    #pragma unroll
    for (int j = 0; j < 16; ++j) {
      unsigned short h, l;
      bf16_split(va[j], h, l);
      ahu[j] = h; alu[j] = l;
    }
    const v16bf ah = __builtin_bit_cast(v16bf, ahu);
    const v16bf al = __builtin_bit_cast(v16bf, alu);

    #pragma unroll
    for (int nt = 0; nt < 8; ++nt) {
      // B 32x16 layout: lane = column (l15), lanes 0-15 K 0..15, lanes 16-31 K 16..31
      const int col  = nt * 16 + l15;
      const int boff = col * 128 + kc * 32 + half * 16;   // 32B-aligned
      const v16bf bh = *(const v16bf *)(BTh + boff);
      const v16bf bl = *(const v16bf *)(BTl + boff);
      acc[nt] = __builtin_amdgcn_wmma_f32_16x16x32_bf16(false, ah, false, bh, (short)0, acc[nt], false, false);
      acc[nt] = __builtin_amdgcn_wmma_f32_16x16x32_bf16(false, ah, false, bl, (short)0, acc[nt], false, false);
      acc[nt] = __builtin_amdgcn_wmma_f32_16x16x32_bf16(false, al, false, bh, (short)0, acc[nt], false, false);
    }
  }

  // C/D f32 layout: VGPR i, lanes 0-15 -> M=i, lanes 16-31 -> M=8+i; N=l15
  #pragma unroll
  for (int nt = 0; nt < 8; ++nt) {
    const int col = nt * 16 + l15;
    #pragma unroll
    for (int i = 0; i < 8; ++i) {
      const int r = row0 + half * 8 + i;
      if (r < nrows) Out[(long)r * 128 + col] = acc[nt][i];
    }
  }
}

// ---------------- message passing ----------------
// agg[n] = dinv[n]^2 * h[n]   (self-loop; also serves as init of agg)
__global__ void k_selfloop(const float *__restrict__ h, const float *__restrict__ dinv,
                           float *__restrict__ agg, int N) {
  int t = blockIdx.x * blockDim.x + threadIdx.x;
  int n = t >> 5, lane = t & 31;
  if (n >= N) return;
  float d = dinv[n], w = d * d;
  float4 v = ((const float4 *)(h + (long)n * 128))[lane];
  ((float4 *)(agg + (long)n * 128))[lane] = make_float4(v.x * w, v.y * w, v.z * w, v.w * w);
}

// one wave per edge: gather h[src], scale by dinv[src]*dinv[dst], atomic-add into agg[dst]
__global__ void k_scatter(const float *__restrict__ h, const float *__restrict__ dinv,
                          const int *__restrict__ src, const int *__restrict__ dst,
                          float *__restrict__ agg, int E) {
  int t = blockIdx.x * blockDim.x + threadIdx.x;
  int e = t >> 5, lane = t & 31;
  if (e >= E) return;
  int s = src[e], d = dst[e];
  float nrm = dinv[s] * dinv[d];
  float4 v = ((const float4 *)(h + (long)s * 128))[lane];
  float *ap = agg + (long)d * 128 + lane * 4;
  atomicAdd(ap + 0, v.x * nrm);
  atomicAdd(ap + 1, v.y * nrm);
  atomicAdd(ap + 2, v.z * nrm);
  atomicAdd(ap + 3, v.w * nrm);
}

// ---------------- batchnorm ----------------
__global__ __launch_bounds__(128) void k_bn_stats(const float *__restrict__ agg,
                                                  float *__restrict__ stats, int N) {
  int c = threadIdx.x, r0 = blockIdx.x * 64;
  float s = 0.f, q = 0.f;
  for (int i = 0; i < 64; ++i) {
    int r = r0 + i;
    if (r < N) { float v = agg[(long)r * 128 + c]; s += v; q += v * v; }
  }
  atomicAdd(&stats[c], s);
  atomicAdd(&stats[128 + c], q);
}

__global__ __launch_bounds__(128) void k_bn_finalize(float *__restrict__ stats,
                                                     const float *__restrict__ gamma,
                                                     const float *__restrict__ beta, int N) {
  int c = threadIdx.x;
  float inv_n = 1.0f / (float)N;
  float mu  = stats[c] * inv_n;
  float var = stats[128 + c] * inv_n - mu * mu;
  float sc  = gamma[c] * rsqrtf(var + BN_EPS);
  stats[c]       = sc;                 // scale
  stats[128 + c] = beta[c] - mu * sc;  // shift (gcn1 bias cancels under mean-subtraction)
}

__global__ void k_bn_act(const float *__restrict__ agg, const float *__restrict__ stats,
                         float *__restrict__ out, int total) {
  int t = blockIdx.x * blockDim.x + threadIdx.x;
  if (t >= total) return;
  int c = t & 127;
  float v = agg[t] * stats[c] + stats[128 + c];
  out[t] = v > 0.f ? v : LEAKY_SLOPE * v;
}

// ---------------- pooling + head ----------------
__global__ __launch_bounds__(128) void k_pool(const float *__restrict__ agg,
                                              const int *__restrict__ batch,
                                              float *__restrict__ gsum, float *__restrict__ gcnt,
                                              int N) {
  int c = threadIdx.x, r0 = blockIdx.x * 64;
  for (int i = 0; i < 64; ++i) {
    int r = r0 + i;
    if (r < N) {
      int g = batch[r];
      atomicAdd(&gsum[(long)g * 128 + c], agg[(long)r * 128 + c]);
      if (c == 0) atomicAdd(&gcnt[g], 1.0f);
    }
  }
}

__global__ __launch_bounds__(128) void k_final(const float *__restrict__ gsum,
                                               const float *__restrict__ gcnt,
                                               const float *__restrict__ b2,
                                               const float *__restrict__ lin_w,
                                               const float *__restrict__ lin_b,
                                               float *__restrict__ out) {
  __shared__ float red[128];
  int g = blockIdx.x, c = threadIdx.x;
  float cnt = gcnt[g]; if (cnt < 1.f) cnt = 1.f;
  float pooled = gsum[(long)g * 128 + c] / cnt + b2[c];   // gcn2 bias folded here
  red[c] = pooled * lin_w[c];
  __syncthreads();
  for (int s = 64; s > 0; s >>= 1) {
    if (c < s) red[c] += red[c + s];
    __syncthreads();
  }
  if (c == 0) out[g] = red[0] + lin_b[0];
}

// ---------------- launcher ----------------
extern "C" void kernel_launch(void *const *d_in, const int *in_sizes, int n_in,
                              void *d_out, int out_size, void *d_ws, size_t ws_size,
                              hipStream_t stream) {
  (void)n_in; (void)ws_size;
  const float *x     = (const float *)d_in[0];
  const float *w1    = (const float *)d_in[1];
  /* d_in[2] = gcn1_b: cancels exactly under batch-norm mean subtraction */
  const float *w2    = (const float *)d_in[3];
  const float *b2    = (const float *)d_in[4];
  const float *gamma = (const float *)d_in[5];
  const float *beta  = (const float *)d_in[6];
  const float *lin_w = (const float *)d_in[7];
  const float *lin_b = (const float *)d_in[8];
  const int   *ei    = (const int *)d_in[9];
  const int   *batch = (const int *)d_in[10];

  const int N = in_sizes[0] / 128;
  const int E = in_sizes[9] / 2;
  const int G = out_size;

  // workspace layout
  char *w = (char *)d_ws;
  float *dinv = (float *)w;              w += (size_t)((N + 63) & ~63) * 4;
  unsigned short *wt1h = (unsigned short *)w; w += 128 * 128 * 2;
  unsigned short *wt1l = (unsigned short *)w; w += 128 * 128 * 2;
  unsigned short *wt2h = (unsigned short *)w; w += 128 * 128 * 2;
  unsigned short *wt2l = (unsigned short *)w; w += 128 * 128 * 2;
  float *bufA  = (float *)w;             w += (size_t)N * 128 * 4;
  float *bufB  = (float *)w;             w += (size_t)N * 128 * 4;
  float *stats = (float *)w;             // 256 + G*128 + G floats, contiguous
  float *gsum  = stats + 256;
  float *gcnt  = gsum + (size_t)G * 128;

  const int statN = 256 + G * 128 + G;
  const int gN32  = (N * 32 + 255) / 256;
  const int gE32  = (E * 32 + 255) / 256;
  const int gGemm = (N + 127) / 128;

  k_zero<<<(N + 255) / 256, 256, 0, stream>>>(dinv, N);
  k_zero<<<(statN + 255) / 256, 256, 0, stream>>>(stats, statN);

  k_degree<<<(E + 255) / 256, 256, 0, stream>>>(ei + E, dinv, E);
  k_dinv<<<(N + 255) / 256, 256, 0, stream>>>(dinv, N);

  k_prep_w<<<64, 256, 0, stream>>>(w1, wt1h, wt1l);
  k_prep_w<<<64, 256, 0, stream>>>(w2, wt2h, wt2l);

  // layer 1
  k_gemm128<<<gGemm, 256, 0, stream>>>(x, wt1h, wt1l, bufA, N);
  k_selfloop<<<gN32, 256, 0, stream>>>(bufA, dinv, bufB, N);
  k_scatter<<<gE32, 256, 0, stream>>>(bufA, dinv, ei, ei + E, bufB, E);

  // batchnorm + leaky relu
  k_bn_stats<<<(N + 63) / 64, 128, 0, stream>>>(bufB, stats, N);
  k_bn_finalize<<<1, 128, 0, stream>>>(stats, gamma, beta, N);
  k_bn_act<<<(N * 128 + 255) / 256, 256, 0, stream>>>(bufB, stats, bufA, N * 128);

  // layer 2
  k_gemm128<<<gGemm, 256, 0, stream>>>(bufA, wt2h, wt2l, bufB, N);
  k_selfloop<<<gN32, 256, 0, stream>>>(bufB, dinv, bufA, N);
  k_scatter<<<gE32, 256, 0, stream>>>(bufB, dinv, ei, ei + E, bufA, E);

  // pool + linear head
  k_pool<<<(N + 63) / 64, 128, 0, stream>>>(bufA, batch, gsum, gcnt, N);
  k_final<<<G, 128, 0, stream>>>(gsum, gcnt, b2, lin_w, lin_b, (float *)d_out);
}
